// Attention_6184752906333
// MI455X (gfx1250) — matrix operand order repaired and checked by reading, not compiled, not run on hardware
//
#include <hip/hip_runtime.h>

#define NUM_HEADS     32
#define HEAD_DIM      128
#define NUM_KV_HEADS  8
#define SEQ_LEN       2048
#define BATCH         2
// Softmax run in base 2: fold log2(e) into the QK scale so probabilities are
// exp2(s - m) == softmax(q.k / sqrt(d)).  v_exp_f32 is natively 2^x.
#define QK_SCALE_L2   (0.08838834764831845f * 1.4426950408889634f)

typedef __attribute__((ext_vector_type(16))) _Float16 v16h;
typedef __attribute__((ext_vector_type(8)))  _Float16 v8h;
typedef __attribute__((ext_vector_type(4)))  _Float16 v4h;
typedef __attribute__((ext_vector_type(8)))  float    v8f;
typedef __attribute__((ext_vector_type(4)))  float    v4f;   // clang vector: OK for nontemporal builtins

union V16 { v16h v; v8h h[2]; };

static __device__ __forceinline__ float fast_exp2(float x) {
#if __has_builtin(__builtin_amdgcn_exp2f)
    return __builtin_amdgcn_exp2f(x);      // v_exp_f32
#else
    return exp2f(x);
#endif
}

// Flash-attention-2, causal GQA.  One workgroup = 64 q-rows of one (batch, head),
// 4 waves x 16 q-rows.  K/V tiles (32 keys x 128d) double-buffered in LDS as f16
// (V transposed at store time for contiguous B-fragment ds_load_b128s).  Both
// GEMMs on v_wmma_f32_16x16x32_f16 with f32 accumulation and f32 softmax state.
// One barrier per key block; next tile's global loads issued before compute.
__global__ __launch_bounds__(128)
void fa_causal_gqa_kernel(const float* __restrict__ Qg,
                          const float* __restrict__ Kg,
                          const float* __restrict__ Vg,
                          float* __restrict__ Og)
{
    // Rows padded by 8 halves (16B) to break the 64-bank x 4B = 256B period.
    __shared__ alignas(16) _Float16 Kl0[32][136];    // [key][d]     8704 B
    __shared__ alignas(16) _Float16 Vt0[128][40];    // [d][key]    10240 B
    __shared__ alignas(16) _Float16 Kl1[32][136];
    __shared__ alignas(16) _Float16 Vt1[128][40];
    __shared__ alignas(16) _Float16 Pl[4][16][40];   // per-wave P   5120 B

    const int tid  = threadIdx.x;
    const int wave = tid >> 5;
    const int lane = tid & 31;
    const int lrow = lane & 15;    // M (A/C) or N (B) index
    const int lgrp = lane >> 4;    // half-wave group

    const int b     = blockIdx.z;
    const int h     = blockIdx.y;
    const int hk    = h >> 2;                  // 32 heads -> 8 kv heads
    const int q0_wg = blockIdx.x * 64;
    const int q0    = q0_wg + wave * 16;

    // ---- Q tile (16 x 128) -> 4 A-fragments (f16, pre-scaled) -------------
    // ISA A-layout (16-bit 16x32): lane = m (+16 per group), half slot j:
    //   K = (j<8 ? j : 8+j) + 8*lgrp   -> two contiguous 8-half chunks.
    // Q rows are touched exactly once across the whole grid -> nontemporal.
    v16h a_q[4];
    const float* qrow = Qg + (size_t)(b * SEQ_LEN + q0 + lrow) * (NUM_HEADS * HEAD_DIM)
                           + h * HEAD_DIM;
#pragma unroll
    for (int c = 0; c < 4; ++c) {
        const v4f* p0 = (const v4f*)(qrow + 32 * c + 8 * lgrp);
        const v4f* p1 = (const v4f*)(qrow + 32 * c + 16 + 8 * lgrp);
        const v4f f0 = __builtin_nontemporal_load(p0);
        const v4f f1 = __builtin_nontemporal_load(p0 + 1);
        const v4f f2 = __builtin_nontemporal_load(p1);
        const v4f f3 = __builtin_nontemporal_load(p1 + 1);
#pragma unroll
        for (int j = 0; j < 4; ++j) {
            a_q[c][j]      = (_Float16)(f0[j] * QK_SCALE_L2);
            a_q[c][4 + j]  = (_Float16)(f1[j] * QK_SCALE_L2);
            a_q[c][8 + j]  = (_Float16)(f2[j] * QK_SCALE_L2);
            a_q[c][12 + j] = (_Float16)(f3[j] * QK_SCALE_L2);
        }
    }

    // O accumulator: 8 d-tiles of 16x16 f32 in C layout (row = r + 8*lgrp).
    v8f o_acc[8];
#pragma unroll
    for (int dt = 0; dt < 8; ++dt)
#pragma unroll
        for (int r = 0; r < 8; ++r) o_acc[dt][r] = 0.0f;

    float m_i[8], l_i[8];
#pragma unroll
    for (int r = 0; r < 8; ++r) { m_i[r] = -1e30f; l_i[r] = 0.0f; }

    // Causal bound for the whole workgroup (uniform -> EXEC stays full at WMMAs).
    const int nkt = 2 * blockIdx.x + 2;   // 32-key blocks up to q0_wg + 64

    // ---- cooperative K/V tile load: 32 keys x 128d, f32 -> f16 ------------
    // Also prefetches the tile after this one into L2 (global_prefetch_b8).
    auto load_tile = [&](int kblk, _Float16 (*Kd)[136], _Float16 (*Vd)[40]) {
        const int k0 = kblk * 32;
        const int kp = ((kblk + 1) * 32 < SEQ_LEN) ? (kblk + 1) * 32 : k0;
#pragma unroll
        for (int i = 0; i < 8; ++i) {
            const int idx = i * 128 + tid;          // 0..1023 float4 slots
            const int key = idx >> 5;               // 0..31
            const int d4  = (idx & 31) << 2;        // 0..124
            const size_t rowoff =
                (size_t)(b * SEQ_LEN + k0 + key) * (NUM_KV_HEADS * HEAD_DIM)
                + hk * HEAD_DIM + d4;
            if ((tid & 3) == 0) {                   // one lane per 64B line
                const size_t poff =
                    (size_t)(b * SEQ_LEN + kp + key) * (NUM_KV_HEADS * HEAD_DIM)
                    + hk * HEAD_DIM + d4;
                __builtin_prefetch(Kg + poff, 0, 1);
                __builtin_prefetch(Vg + poff, 0, 1);
            }
            const v4f kk = *(const v4f*)(Kg + rowoff);
            v4h kh;
            kh[0] = (_Float16)kk[0]; kh[1] = (_Float16)kk[1];
            kh[2] = (_Float16)kk[2]; kh[3] = (_Float16)kk[3];
            *(v4h*)&Kd[key][d4] = kh;
            const v4f vv = *(const v4f*)(Vg + rowoff);
            Vd[d4 + 0][key] = (_Float16)vv[0];
            Vd[d4 + 1][key] = (_Float16)vv[1];
            Vd[d4 + 2][key] = (_Float16)vv[2];
            Vd[d4 + 3][key] = (_Float16)vv[3];
        }
    };

    // ---- per-wave compute on one staged 32-key tile -----------------------
    auto compute_tile = [&](int k0, const _Float16 (*Kd)[136],
                            const _Float16 (*Vd)[40]) {
        // S = Q K^T : two 16x16 score tiles over 32 keys.
        // B-layout (32x16): lane n = lane%16, K = (j<8 ? j : 8+j) + 8*lgrp (matches A).
        v8f c0, c1;
#pragma unroll
        for (int r = 0; r < 8; ++r) { c0[r] = 0.0f; c1[r] = 0.0f; }
#pragma unroll
        for (int c = 0; c < 4; ++c) {
            V16 b0, b1;
            const _Float16* kr0 = &Kd[lrow][32 * c + 8 * lgrp];
            const _Float16* kr1 = &Kd[16 + lrow][32 * c + 8 * lgrp];
            b0.h[0] = *(const v8h*)(kr0);
            b0.h[1] = *(const v8h*)(kr0 + 16);
            b1.h[0] = *(const v8h*)(kr1);
            b1.h[1] = *(const v8h*)(kr1 + 16);
            c0 = __builtin_amdgcn_wmma_f32_16x16x32_f16(false, a_q[c], false, b0.v,
                                                        (short)0, c0, false, false);
            c1 = __builtin_amdgcn_wmma_f32_16x16x32_f16(false, a_q[c], false, b1.v,
                                                        (short)0, c1, false, false);
        }

        // Causal mask + online softmax, base 2 (rows r+8*lgrp, cols lrow / lrow+16).
        const int row_base = q0 + 8 * lgrp;
        const int col0 = k0 + lrow;
        const int col1 = col0 + 16;
        float alpha[8], p0v[8], p1v[8];
#pragma unroll
        for (int r = 0; r < 8; ++r) {
            const int row = row_base + r;
            const float s0 = (col0 <= row) ? c0[r] : -1e30f;
            const float s1 = (col1 <= row) ? c1[r] : -1e30f;
            float t = fmaxf(s0, s1);                  // row-max across 16 lanes
            t = fmaxf(t, __shfl_xor(t, 1, 16));
            t = fmaxf(t, __shfl_xor(t, 2, 16));
            t = fmaxf(t, __shfl_xor(t, 4, 16));
            t = fmaxf(t, __shfl_xor(t, 8, 16));
            const float mnew = fmaxf(m_i[r], t);
            const float a  = fast_exp2(m_i[r] - mnew);
            const float p0 = fast_exp2(s0 - mnew);
            const float p1 = fast_exp2(s1 - mnew);
            float rs = p0 + p1;                       // row-sum across 16 lanes
            rs += __shfl_xor(rs, 1, 16);
            rs += __shfl_xor(rs, 2, 16);
            rs += __shfl_xor(rs, 4, 16);
            rs += __shfl_xor(rs, 8, 16);
            m_i[r]  = mnew;
            l_i[r]  = a * l_i[r] + rs;
            alpha[r] = a;
            p0v[r] = p0; p1v[r] = p1;
        }

        // Rescale running output.
#pragma unroll
        for (int dt = 0; dt < 8; ++dt)
#pragma unroll
            for (int r = 0; r < 8; ++r) o_acc[dt][r] *= alpha[r];

        // P: C-layout -> per-wave LDS -> A-fragment (same-wave DS is in-order).
#pragma unroll
        for (int r = 0; r < 8; ++r) {
            Pl[wave][8 * lgrp + r][lrow]      = (_Float16)p0v[r];
            Pl[wave][8 * lgrp + r][16 + lrow] = (_Float16)p1v[r];
        }
        V16 ap;
        const _Float16* pr = &Pl[wave][lrow][0];
        ap.h[0] = *(const v8h*)(pr + 8 * lgrp);
        ap.h[1] = *(const v8h*)(pr + 16 + 8 * lgrp);

        // O += P V : 8 WMMAs over the 128-wide head dim.
#pragma unroll
        for (int dt = 0; dt < 8; ++dt) {
            V16 bv;
            const _Float16* vr = &Vd[16 * dt + lrow][8 * lgrp];
            bv.h[0] = *(const v8h*)(vr);
            bv.h[1] = *(const v8h*)(vr + 16);
            o_acc[dt] = __builtin_amdgcn_wmma_f32_16x16x32_f16(false, ap.v, false, bv.v,
                                                               (short)0, o_acc[dt],
                                                               false, false);
        }
    };

    // ---- main loop: double-buffered, one barrier per key block ------------
    // Tile kt lives in buffer (kt & 1).  The barrier at the top of iteration
    // kt guarantees (a) tile kt's LDS stores completed (dscnt==0 before
    // s_barrier_signal) and (b) all waves finished reading the buffer that
    // iteration kt's loader will overwrite.
    load_tile(0, Kl0, Vt0);
    for (int kt = 0; kt < nkt; ++kt) {
        __syncthreads();
        const int nxt = kt + 1;
        if ((kt & 1) == 0) {
            if (nxt < nkt) load_tile(nxt, Kl1, Vt1);
            compute_tile(kt * 32, Kl0, Vt0);
        } else {
            if (nxt < nkt) load_tile(nxt, Kl0, Vt0);
            compute_tile(kt * 32, Kl1, Vt1);
        }
    }

    // ---- epilogue: normalize and store (streamed, bypass L2 retention) ----
    float rl[8];
#pragma unroll
    for (int r = 0; r < 8; ++r) rl[r] = 1.0f / l_i[r];
#pragma unroll
    for (int dt = 0; dt < 8; ++dt)
#pragma unroll
        for (int r = 0; r < 8; ++r) {
            const int row = q0 + 8 * lgrp + r;
            float* dst = Og + (size_t)(b * SEQ_LEN + row) * (NUM_HEADS * HEAD_DIM)
                            + h * HEAD_DIM + 16 * dt + lrow;
            __builtin_nontemporal_store(o_acc[dt][r] * rl[r], dst);
        }
}

extern "C" void kernel_launch(void* const* d_in, const int* in_sizes, int n_in,
                              void* d_out, int out_size, void* d_ws, size_t ws_size,
                              hipStream_t stream) {
    (void)in_sizes; (void)n_in; (void)d_ws; (void)ws_size; (void)out_size;
    const float* q = (const float*)d_in[0];
    const float* k = (const float*)d_in[1];
    const float* v = (const float*)d_in[2];
    float* out = (float*)d_out;
    dim3 grid(SEQ_LEN / 64, NUM_HEADS, BATCH);
    fa_causal_gqa_kernel<<<grid, dim3(128), 0, stream>>>(q, k, v, out);
}
